// Attention_72499047957038
// MI455X (gfx1250) — matrix operand order repaired and checked by reading, not compiled, not run on hardware
//
#include <hip/hip_runtime.h>
#include <hip/hip_bf16.h>
#include <math.h>

// ---- problem constants (from reference) ----
#define HIDDEN  2048
#define NHEADS  32
#define NKV     8
#define HEADDIM 64
#define SEQ     2048
#define BATCH   2
#define TOKENS  (BATCH * SEQ)        // 4096
#define KVDIM   (NKV * HEADDIM)      // 512

typedef __attribute__((ext_vector_type(16))) _Float16 v16h;
typedef __attribute__((ext_vector_type(8)))  _Float16 v8h;
typedef __attribute__((ext_vector_type(8)))  float    v8f;

__device__ __forceinline__ v16h make_frag(v8h lo, v8h hi) {
    v16h r;
#pragma unroll
    for (int i = 0; i < 8; ++i) { r[i] = lo[i]; r[i + 8] = hi[i]; }
    return r;
}

__device__ __forceinline__ v8f wmma_f16(v16h a, v16h b, v8f c) {
    return __builtin_amdgcn_wmma_f32_16x16x32_f16(false, a, false, b, (short)0, c,
                                                  false, false);
}

// gfx1250 async copy global -> LDS (16B per lane), tracked by ASYNCcnt
__device__ __forceinline__ void async_b128(unsigned lds_byte_addr, const void* gaddr) {
    asm volatile("global_load_async_to_lds_b128 %0, %1, off"
                 :: "v"(lds_byte_addr), "v"((unsigned long long)(size_t)gaddr)
                 : "memory");
}
__device__ __forceinline__ void wait_async0() {
    asm volatile("s_wait_asynccnt 0x0" ::: "memory");
}

// ---------------------------------------------------------------------------
// f32 -> f16 convert (plain, for X / activations)
// ---------------------------------------------------------------------------
__global__ void f32_to_f16(const float* __restrict__ in, _Float16* __restrict__ out, int n) {
    int i = blockIdx.x * blockDim.x + threadIdx.x;
    if (i < n) out[i] = (_Float16)in[i];
}

// ---------------------------------------------------------------------------
// f32 [K][N] -> f16 [N][K] (transposed weights: makes GEMM B-staging contiguous)
// ---------------------------------------------------------------------------
__global__ __launch_bounds__(256)
void f32_to_f16_transpose(const float* __restrict__ in, _Float16* __restrict__ out,
                          int K, int N) {
    __shared__ float tile[32][33];
    int kb = blockIdx.y * 32, nb = blockIdx.x * 32;
    int tx = threadIdx.x & 31, ty = threadIdx.x >> 5;
#pragma unroll
    for (int r = ty; r < 32; r += 8)
        tile[r][tx] = in[(size_t)(kb + r) * N + nb + tx];
    __syncthreads();
#pragma unroll
    for (int r = ty; r < 32; r += 8)
        out[(size_t)(nb + r) * K + kb + tx] = (_Float16)tile[tx][r];
}

// ---------------------------------------------------------------------------
// Tiled WMMA GEMM: C[M,N](f32) = A[M,K](f16 rm) * Bt[N,K](f16 rm, = B^T)
// 128x128 block tile, BK=64, 256 threads = 8 waves (4x2), 16 WMMA/wave/step.
// Double-buffered LDS filled with global_load_async_to_lds_b128.
// Per K-chunk: batch-load all 6 fragments (12 ds_load_b128 in flight), then
// issue 8 back-to-back WMMAs — one batched s_wait_dscnt instead of four.
// ---------------------------------------------------------------------------
#define GSTRIDE 72                    // LDS row stride in halfs (144B, 16B-aligned)
#define GBUF    (128 * GSTRIDE)      // halfs per buffer

__global__ __launch_bounds__(256)
void wmma_gemm_f16f32(const _Float16* __restrict__ A, const _Float16* __restrict__ Bt,
                      float* __restrict__ C, int M, int N, int K) {
    __shared__ _Float16 As[2 * GBUF];
    __shared__ _Float16 Bs[2 * GBUF];

    const int tid  = threadIdx.x;
    const int lane = tid & 31;
    const int wave = tid >> 5;
    const int h    = lane >> 4;
    const int ln   = lane & 15;
    const int wr   = wave & 3;        // 4 row tiles of 32
    const int wc   = wave >> 2;       // 2 col tiles of 64
    const int bm   = blockIdx.y * 128;
    const int bn   = blockIdx.x * 128;
    const unsigned as_base = (unsigned)(size_t)&As[0];
    const unsigned bs_base = (unsigned)(size_t)&Bs[0];

    v8f acc[8];
#pragma unroll
    for (int t = 0; t < 8; ++t)
#pragma unroll
        for (int i = 0; i < 8; ++i) acc[t][i] = 0.0f;

    const int nsteps = K >> 6;

    // prologue: async-stage k-slab 0 into buffer 0 (128 rows x 64 halfs each)
#pragma unroll
    for (int j = 0; j < 4; ++j) {
        int ci = tid + 256 * j;
        int row = ci >> 2, kc = (ci & 3) * 16;
        unsigned loff = (unsigned)(row * GSTRIDE + kc) * 2;
        async_b128(as_base + loff, A  + (size_t)(bm + row) * K + kc);
        async_b128(bs_base + loff, Bt + (size_t)(bn + row) * K + kc);
    }

    for (int t = 0; t < nsteps; ++t) {
        wait_async0();            // own async stores for buffer t complete
        __syncthreads();          // everyone's complete; prev reads of t+1 buf done

        if (t + 1 < nsteps) {     // prefetch next slab into the other buffer
            int k0 = (t + 1) << 6;
            unsigned boff = (unsigned)(((t + 1) & 1) * GBUF * 2);
#pragma unroll
            for (int j = 0; j < 4; ++j) {
                int ci = tid + 256 * j;
                int row = ci >> 2, kc = (ci & 3) * 16;
                unsigned loff = boff + (unsigned)(row * GSTRIDE + kc) * 2;
                async_b128(as_base + loff, A  + (size_t)(bm + row) * K + k0 + kc);
                async_b128(bs_base + loff, Bt + (size_t)(bn + row) * K + k0 + kc);
            }
        }

        const _Float16* Ab = &As[(t & 1) * GBUF];
        const _Float16* Bb = &Bs[(t & 1) * GBUF];
#pragma unroll
        for (int c = 0; c < 2; ++c) {
            // batch all fragment loads for this K-chunk before any WMMA
            v16h af[2], bf[4];
#pragma unroll
            for (int mt = 0; mt < 2; ++mt) {
                const _Float16* ap = Ab + (wr * 32 + mt * 16 + ln) * GSTRIDE;
                af[mt] = make_frag(*(const v8h*)(ap + 32 * c + 8 * h),
                                   *(const v8h*)(ap + 32 * c + 16 + 8 * h));
            }
#pragma unroll
            for (int nt = 0; nt < 4; ++nt) {
                const _Float16* bp = Bb + (wc * 64 + nt * 16 + ln) * GSTRIDE;
                bf[nt] = make_frag(*(const v8h*)(bp + 32 * c + 8 * h),
                                   *(const v8h*)(bp + 32 * c + 16 + 8 * h));
            }
#pragma unroll
            for (int nt = 0; nt < 4; ++nt) {
                acc[nt]     = wmma_f16(af[0], bf[nt], acc[nt]);
                acc[4 + nt] = wmma_f16(af[1], bf[nt], acc[4 + nt]);
            }
        }
    }

    // C layout: VGPR r -> row r+8h, col = ln
#pragma unroll
    for (int mt = 0; mt < 2; ++mt)
#pragma unroll
        for (int nt = 0; nt < 4; ++nt)
#pragma unroll
            for (int r = 0; r < 8; ++r) {
                int row = bm + wr * 32 + mt * 16 + r + 8 * h;
                int col = bn + wc * 64 + nt * 16 + ln;
                C[(size_t)row * N + col] = acc[mt * 4 + nt][r];
            }
}

// ---------------------------------------------------------------------------
// RoPE + repack to head-major f16 [b][H][s][d]
// ---------------------------------------------------------------------------
__global__ void rope_pack_q(const float* __restrict__ Q, const int* __restrict__ pos,
                            _Float16* __restrict__ Qh) {
    int tid  = blockIdx.x * blockDim.x + threadIdx.x;
    int i    = tid & 31;
    int head = (tid >> 5) & (NHEADS - 1);
    int tok  = tid >> 10;
    if (tok >= TOKENS) return;
    int bi = tok / SEQ, si = tok % SEQ;
    float p    = (float)pos[tok];
    float invf = __expf(-9.210340371976184f * ((float)(2 * i) * (1.0f / 64.0f)));
    float sv, cv;
    __sincosf(p * invf, &sv, &cv);
    float q0v = Q[(size_t)tok * HIDDEN + head * 64 + i];
    float q1v = Q[(size_t)tok * HIDDEN + head * 64 + i + 32];
    size_t base = (((size_t)bi * NHEADS + head) * SEQ + si) * 64;
    Qh[base + i]      = (_Float16)(q0v * cv - q1v * sv);
    Qh[base + i + 32] = (_Float16)(q1v * cv + q0v * sv);
}

__global__ void rope_pack_k(const float* __restrict__ Kx, const int* __restrict__ pos,
                            _Float16* __restrict__ Kh) {
    int tid  = blockIdx.x * blockDim.x + threadIdx.x;
    int i    = tid & 31;
    int head = (tid >> 5) & (NKV - 1);
    int tok  = tid >> 8;
    if (tok >= TOKENS) return;
    int bi = tok / SEQ, si = tok % SEQ;
    float p    = (float)pos[tok];
    float invf = __expf(-9.210340371976184f * ((float)(2 * i) * (1.0f / 64.0f)));
    float sv, cv;
    __sincosf(p * invf, &sv, &cv);
    float k0v = Kx[(size_t)tok * KVDIM + head * 64 + i];
    float k1v = Kx[(size_t)tok * KVDIM + head * 64 + i + 32];
    size_t base = (((size_t)bi * NKV + head) * SEQ + si) * 64;
    Kh[base + i]      = (_Float16)(k0v * cv - k1v * sv);
    Kh[base + i + 32] = (_Float16)(k1v * cv + k0v * sv);
}

// V packed fully transposed: [b][Hkv][d][s] so attention V-tiles stage contiguously
__global__ void pack_v_t(const float* __restrict__ Vx, _Float16* __restrict__ Vt) {
    int tid  = blockIdx.x * blockDim.x + threadIdx.x;
    int si   = tid & (SEQ - 1);
    int d    = (tid >> 11) & 63;
    int head = (tid >> 17) & (NKV - 1);
    int bi   = tid >> 20;
    if (bi >= BATCH) return;
    float v = Vx[((size_t)(bi * SEQ + si)) * KVDIM + head * 64 + d];
    Vt[(((size_t)bi * NKV + head) * 64 + d) * SEQ + si] = (_Float16)v;
}

// ---------------------------------------------------------------------------
// Flash attention (causal, GQA 4:1). Block = (64 q-rows, head, batch),
// 128 threads = 4 waves. Double-buffered async K/V tile staging.
// ---------------------------------------------------------------------------
#define FSTRIDE 72
#define FBUF    (64 * FSTRIDE)

__global__ __launch_bounds__(128)
void flash_attn(const _Float16* __restrict__ Qh, const _Float16* __restrict__ Kh,
                const _Float16* __restrict__ Vtg, _Float16* __restrict__ Oh) {
    __shared__ _Float16 Ks[2 * FBUF];     // [buf][key][d]   (== B^T layout for Q*K^T)
    __shared__ _Float16 Vs[2 * FBUF];     // [buf][d][key]   (== B^T layout for P*V)
    __shared__ _Float16 Ps[4 * 16 * FSTRIDE];  // per-wave P bounce

    const int tid  = threadIdx.x;
    const int lane = tid & 31;
    const int wave = tid >> 5;
    const int h    = lane >> 4;
    const int ln   = lane & 15;
    const int q0   = blockIdx.x * 64;
    const int head = blockIdx.y;
    const int bi   = blockIdx.z;
    const int kvh  = head / (NHEADS / NKV);
    const unsigned ks_base = (unsigned)(size_t)&Ks[0];
    const unsigned vs_base = (unsigned)(size_t)&Vs[0];

    const size_t kbase = (((size_t)bi * NKV + kvh) * SEQ) * HEADDIM;  // halfs
    const size_t vbase = (((size_t)bi * NKV + kvh) * HEADDIM) * SEQ;  // [d][s]

    // Q fragments held in registers for the whole block
    const int qrow = q0 + wave * 16 + ln;
    const _Float16* qptr = Qh + (((size_t)bi * NHEADS + head) * SEQ + qrow) * HEADDIM;
    v16h qf[2];
#pragma unroll
    for (int c = 0; c < 2; ++c)
        qf[c] = make_frag(*(const v8h*)(qptr + 32 * c + 8 * h),
                          *(const v8h*)(qptr + 32 * c + 16 + 8 * h));

    v8f o[4];
    float m[8], l[8];
#pragma unroll
    for (int nt = 0; nt < 4; ++nt)
#pragma unroll
        for (int i = 0; i < 8; ++i) o[nt][i] = 0.0f;
#pragma unroll
    for (int r = 0; r < 8; ++r) { m[r] = -1e30f; l[r] = 0.0f; }

    const int wave_rmax = q0 + wave * 16 + 15;
    const int nkv = blockIdx.x + 1;

    // prologue: stage kv tile 0 into buffer 0 (2 x 256 16B chunks, 128 threads)
#pragma unroll
    for (int j = 0; j < 2; ++j) {
        int ci = tid + 128 * j;
        int row = ci >> 2, kc = (ci & 3) * 16;
        unsigned loff = (unsigned)(row * FSTRIDE + kc) * 2;
        async_b128(ks_base + loff, Kh  + kbase + (size_t)row * HEADDIM + kc);
        async_b128(vs_base + loff, Vtg + vbase + (size_t)row * SEQ + kc);
    }

    for (int t = 0; t < nkv; ++t) {
        const int kv0 = t * 64;
        wait_async0();
        __syncthreads();

        if (t + 1 < nkv) {
            int kn = (t + 1) * 64;
            unsigned boff = (unsigned)(((t + 1) & 1) * FBUF * 2);
#pragma unroll
            for (int j = 0; j < 2; ++j) {
                int ci = tid + 128 * j;
                int row = ci >> 2, kc = (ci & 3) * 16;
                unsigned loff = boff + (unsigned)(row * FSTRIDE + kc) * 2;
                async_b128(ks_base + loff, Kh  + kbase + (size_t)(kn + row) * HEADDIM + kc);
                async_b128(vs_base + loff, Vtg + vbase + (size_t)row * SEQ + kn + kc);
            }
        }

        if (kv0 > wave_rmax) continue;   // whole tile masked for this wave

        const _Float16* Kb = &Ks[(t & 1) * FBUF];
        const _Float16* Vb = &Vs[(t & 1) * FBUF];

        // S = Q * K^T
        v8f s[4];
#pragma unroll
        for (int nt = 0; nt < 4; ++nt) {
            v8f a;
#pragma unroll
            for (int i = 0; i < 8; ++i) a[i] = 0.0f;
            const _Float16* bp = Kb + (nt * 16 + ln) * FSTRIDE;
#pragma unroll
            for (int c = 0; c < 2; ++c) {
                v16h bf = make_frag(*(const v8h*)(bp + 32 * c + 8 * h),
                                    *(const v8h*)(bp + 32 * c + 16 + 8 * h));
                a = wmma_f16(qf[c], bf, a);
            }
            s[nt] = a;
        }

        // scale + causal mask + running row max
        const float scale = 0.125f;
        float mnew[8];
#pragma unroll
        for (int r = 0; r < 8; ++r) {
            int row = q0 + wave * 16 + r + 8 * h;
            float mx = m[r];
#pragma unroll
            for (int nt = 0; nt < 4; ++nt) {
                int col = kv0 + nt * 16 + ln;
                float v = s[nt][r] * scale;
                v = (col <= row) ? v : -1e30f;
                s[nt][r] = v;
                mx = fmaxf(mx, v);
            }
#pragma unroll
            for (int off = 1; off < 16; off <<= 1)
                mx = fmaxf(mx, __shfl_xor(mx, off, 32));
            mnew[r] = mx;
        }
#pragma unroll
        for (int r = 0; r < 8; ++r) {
            float alpha = __expf(m[r] - mnew[r]);
            m[r] = mnew[r];
            l[r] *= alpha;
#pragma unroll
            for (int nt = 0; nt < 4; ++nt) o[nt][r] *= alpha;
        }

        // P = exp(S - m); bounce through wave-private LDS (C-layout -> A-layout)
        float psum[8];
#pragma unroll
        for (int r = 0; r < 8; ++r) psum[r] = 0.0f;
        _Float16* pw = &Ps[wave * 16 * FSTRIDE];
#pragma unroll
        for (int nt = 0; nt < 4; ++nt)
#pragma unroll
            for (int r = 0; r < 8; ++r) {
                float p = __expf(s[nt][r] - m[r]);
                pw[(r + 8 * h) * FSTRIDE + nt * 16 + ln] = (_Float16)p;
                psum[r] += p;
            }
#pragma unroll
        for (int r = 0; r < 8; ++r) {
#pragma unroll
            for (int off = 1; off < 16; off <<= 1)
                psum[r] += __shfl_xor(psum[r], off, 32);
            l[r] += psum[r];
        }

        asm volatile("s_wait_dscnt 0" ::: "memory");  // wave-private region

        v16h pf[2];
#pragma unroll
        for (int c = 0; c < 2; ++c)
            pf[c] = make_frag(*(const v8h*)(pw + ln * FSTRIDE + 32 * c + 8 * h),
                              *(const v8h*)(pw + ln * FSTRIDE + 32 * c + 16 + 8 * h));

        // O += P * V
#pragma unroll
        for (int nt = 0; nt < 4; ++nt) {
            const _Float16* vp = Vb + (nt * 16 + ln) * FSTRIDE;
#pragma unroll
            for (int c = 0; c < 2; ++c) {
                v16h vf = make_frag(*(const v8h*)(vp + 32 * c + 8 * h),
                                    *(const v8h*)(vp + 32 * c + 16 + 8 * h));
                o[nt] = wmma_f16(pf[c], vf, o[nt]);
            }
        }
    }

    // normalize, emit token-major f16 [b][s][H*d]
#pragma unroll
    for (int nt = 0; nt < 4; ++nt)
#pragma unroll
        for (int r = 0; r < 8; ++r) {
            int row = q0 + wave * 16 + r + 8 * h;
            float val = o[nt][r] / l[r];
            Oh[((size_t)bi * SEQ + row) * HIDDEN + head * HEADDIM + nt * 16 + ln] =
                (_Float16)val;
        }
}

// ---------------------------------------------------------------------------
// host launch  (workspace need: ~113.3 MB)
// ---------------------------------------------------------------------------
extern "C" void kernel_launch(void* const* d_in, const int* in_sizes, int n_in,
                              void* d_out, int out_size, void* d_ws, size_t ws_size,
                              hipStream_t stream) {
    (void)in_sizes; (void)n_in; (void)out_size; (void)ws_size;
    const float* X  = (const float*)d_in[0];
    const float* Wq = (const float*)d_in[1];
    const float* Wk = (const float*)d_in[2];
    const float* Wv = (const float*)d_in[3];
    const float* Wo = (const float*)d_in[4];
    const int*  pos = (const int*)d_in[5];
    float* out = (float*)d_out;

    char* ws = (char*)d_ws;
    size_t o = 0;
    _Float16* Xh   = (_Float16*)(ws + o); o += (size_t)TOKENS * HIDDEN * 2;
    _Float16* WqT  = (_Float16*)(ws + o); o += (size_t)HIDDEN * HIDDEN * 2;  // [N][K]
    _Float16* WkT  = (_Float16*)(ws + o); o += (size_t)HIDDEN * KVDIM * 2;
    _Float16* WvT  = (_Float16*)(ws + o); o += (size_t)HIDDEN * KVDIM * 2;
    _Float16* WoT  = (_Float16*)(ws + o); o += (size_t)HIDDEN * HIDDEN * 2;
    size_t q32_off = o;
    float* Q32 = (float*)(ws + o); o += (size_t)TOKENS * HIDDEN * 4;
    float* K32 = (float*)(ws + o); o += (size_t)TOKENS * KVDIM * 4;
    float* V32 = (float*)(ws + o); o += (size_t)TOKENS * KVDIM * 4;
    _Float16* Qh  = (_Float16*)(ws + o); o += (size_t)TOKENS * HIDDEN * 2;
    _Float16* Kh  = (_Float16*)(ws + o); o += (size_t)TOKENS * KVDIM * 2;
    _Float16* Vtg = (_Float16*)(ws + o); o += (size_t)TOKENS * KVDIM * 2;
    _Float16* Ah  = (_Float16*)(ws + q32_off);   // reuse Q32 region

    // conversions (weights transposed so GEMM B-staging is contiguous)
    f32_to_f16<<<(TOKENS * HIDDEN + 255) / 256, 256, 0, stream>>>(X, Xh, TOKENS * HIDDEN);
    f32_to_f16_transpose<<<dim3(HIDDEN / 32, HIDDEN / 32), 256, 0, stream>>>(
        Wq, WqT, HIDDEN, HIDDEN);
    f32_to_f16_transpose<<<dim3(KVDIM / 32, HIDDEN / 32), 256, 0, stream>>>(
        Wk, WkT, HIDDEN, KVDIM);
    f32_to_f16_transpose<<<dim3(KVDIM / 32, HIDDEN / 32), 256, 0, stream>>>(
        Wv, WvT, HIDDEN, KVDIM);
    f32_to_f16_transpose<<<dim3(HIDDEN / 32, HIDDEN / 32), 256, 0, stream>>>(
        Wo, WoT, HIDDEN, HIDDEN);

    // projections
    wmma_gemm_f16f32<<<dim3(HIDDEN / 128, TOKENS / 128), 256, 0, stream>>>(
        Xh, WqT, Q32, TOKENS, HIDDEN, HIDDEN);
    wmma_gemm_f16f32<<<dim3(KVDIM / 128, TOKENS / 128), 256, 0, stream>>>(
        Xh, WkT, K32, TOKENS, KVDIM, HIDDEN);
    wmma_gemm_f16f32<<<dim3(KVDIM / 128, TOKENS / 128), 256, 0, stream>>>(
        Xh, WvT, V32, TOKENS, KVDIM, HIDDEN);

    // RoPE + repack
    rope_pack_q<<<(TOKENS * NHEADS * 32) / 256, 256, 0, stream>>>(Q32, pos, Qh);
    rope_pack_k<<<(TOKENS * NKV * 32) / 256, 256, 0, stream>>>(K32, pos, Kh);
    pack_v_t<<<(TOKENS * NKV * 64) / 256, 256, 0, stream>>>(V32, Vtg);

    // causal GQA attention
    flash_attn<<<dim3(SEQ / 64, NHEADS, BATCH), 128, 0, stream>>>(Qh, Kh, Vtg, Ah);

    // output projection
    wmma_gemm_f16f32<<<dim3(HIDDEN / 128, TOKENS / 128), 256, 0, stream>>>(
        Ah, WoT, out, TOKENS, HIDDEN, HIDDEN);
}
